// DDSCTransformer_71073118814214
// MI455X (gfx1250) — matrix operand order repaired and checked by reading, not compiled, not run on hardware
//
#include <hip/hip_runtime.h>
#include <hip/hip_bf16.h>
#include <math.h>
#include <stdint.h>

typedef _Float16 h16;
typedef __attribute__((ext_vector_type(16))) _Float16 v16h;
typedef __attribute__((ext_vector_type(8)))  _Float16 v8h;
typedef __attribute__((ext_vector_type(8)))  float    v8f;
typedef int i32x4 __attribute__((vector_size(16)));

#define B_  16
#define C_  64
#define L_  1024
#define H_  8
#define HC_ 512
#define LN_EPS 1e-5f

#if defined(__has_builtin)
#  if __has_builtin(__builtin_amdgcn_global_load_async_to_lds_b128)
#    define HAVE_ASYNC_LDS 1
#  endif
#endif

__device__ __forceinline__ v8f wmma16(v16h a, v16h b, v8f c) {
  return __builtin_amdgcn_wmma_f32_16x16x32_f16(false, a, false, b, (short)0, c,
                                                false, false);
}

// A-fragment (16x32 f16): lane t holds row M=t%16; halves 0..15 = K of ksel*16..ksel*16+15
// -> two 16B contiguous groups.
__device__ __forceinline__ v16h a_frag_f16(const h16* plo, const h16* phi) {
  union { v16h v; v8h h[2]; } u;
  u.h[0] = *(const v8h*)plo;
  u.h[1] = *(const v8h*)phi;
  return u.v;
}

__device__ __forceinline__ unsigned pack2(float a, float b) {
  union { h16 h[2]; unsigned u; } p;
  p.h[0] = (h16)a; p.h[1] = (h16)b;
  return p.u;
}

// 16B global -> LDS copy: async on gfx1250 when the builtin exists.
__device__ __forceinline__ void cp16(const h16* g, h16* l) {
#ifdef HAVE_ASYNC_LDS
  __builtin_amdgcn_global_load_async_to_lds_b128(
      (__attribute__((address_space(1))) i32x4*)(uintptr_t)g,
      (__attribute__((address_space(3))) i32x4*)(uintptr_t)l, 0, 0);
#else
  *(v8h*)l = *(const v8h*)g;
#endif
}

template <int N>
__device__ __forceinline__ void wait_async_le() {
#ifdef HAVE_ASYNC_LDS
#  if __has_builtin(__builtin_amdgcn_s_wait_asynccnt)
  __builtin_amdgcn_s_wait_asynccnt((short)N);
#  else
  asm volatile("s_wait_asynccnt %0" ::"i"(N) : "memory");
#  endif
#endif
}

// ---------------- K0: q (b,c,l) f32 -> xqT (b,l,c) f16 ----------------
__global__ void k0_transpose(const float* __restrict__ q, h16* __restrict__ xqT) {
  int idx = blockIdx.x * blockDim.x + threadIdx.x;   // B*C*L = 1M
  int b = idx >> 16;           // C*L = 65536
  int c = (idx >> 10) & 63;
  int l = idx & 1023;
  xqT[((size_t)b * L_ + l) * C_ + c] = (h16)q[idx];
}

// ---------------- Kcvt: f32 weights -> f16 (done once, reused everywhere) ----
__global__ void kcvt(const float* __restrict__ src, h16* __restrict__ dst, int n) {
  int i = blockIdx.x * blockDim.x + threadIdx.x;
  if (i < n) dst[i] = (h16)src[i];
}

// ---------------- K1: fused QKV pointwise GEMM + gated depthwise conv --------
#define TL_ 128
__global__ __launch_bounds__(32) void k1_qkv(
    const h16* __restrict__ xqT,
    const h16* __restrict__ pwq, const float* __restrict__ gq,
    const float* __restrict__ d3q, const float* __restrict__ d15q,
    const h16* __restrict__ pwk, const float* __restrict__ gk,
    const float* __restrict__ d3k, const float* __restrict__ d15k,
    const h16* __restrict__ pwv, const float* __restrict__ gv,
    const float* __restrict__ d3v, const float* __restrict__ d15v,
    h16* __restrict__ qbuf, h16* __restrict__ kbuf, h16* __restrict__ vbuf) {
  __shared__ float sm[16][TL_ + 20];
  int t  = threadIdx.x;
  int l0 = blockIdx.x * TL_;
  int o0 = blockIdx.y * 16;
  int bz = blockIdx.z;
  int b  = bz / 3, p = bz % 3;
  const h16*   pw   = (p == 0) ? pwq  : (p == 1) ? pwk  : pwv;
  const float* gt   = (p == 0) ? gq   : (p == 1) ? gk   : gv;
  const float* dw3  = (p == 0) ? d3q  : (p == 1) ? d3k  : d3v;
  const float* dw15 = (p == 0) ? d15q : (p == 1) ? d15k : d15v;

  int m = t & 15, ks = t >> 4;
  const h16* wr = pw + (size_t)(o0 + m) * C_;
  v16h A0 = a_frag_f16(wr + ks * 16,      wr + 8 + ks * 16);
  v16h A1 = a_frag_f16(wr + 32 + ks * 16, wr + 40 + ks * 16);

  // 9 column sub-tiles cover [l0-8, l0+136) so the conv halo (+-7) is in LDS
  for (int j = 0; j < 9; ++j) {
    int l = l0 - 8 + j * 16 + m;
    v16h B0 = {}, B1 = {};
    if (l >= 0 && l < L_) {
      const h16* xp = xqT + ((size_t)b * L_ + l) * C_ + ks * 16;
      B0 = *(const v16h*)xp;
      B1 = *(const v16h*)(xp + 32);
    }
    v8f acc = {};
    acc = wmma16(A0, B0, acc);
    acc = wmma16(A1, B1, acc);
#pragma unroll
    for (int r = 0; r < 8; ++r)
      sm[r + 8 * ks][j * 16 + m] = acc[r];
  }
  __syncthreads();

  float e0 = __expf(gt[0]), e1 = __expf(gt[1]);
  float ga = e0 / (e0 + e1), gb = e1 / (e0 + e1);
  float osc = (p == 0) ? 0.125f : 1.0f;   // fold c^-0.5 attention scale into Q

  int h = o0 >> 6;
  int cbase = o0 & 63;
  for (int row = 0; row < 16; ++row) {
    int o = o0 + row;
    float w3r[3], w15r[15];
#pragma unroll
    for (int d = 0; d < 3; ++d)  w3r[d]  = dw3[o * 3 + d];
#pragma unroll
    for (int d = 0; d < 15; ++d) w15r[d] = dw15[o * 15 + d];
#pragma unroll
    for (int cb = 0; cb < TL_ / 32; ++cb) {
      int lc = t + 32 * cb;
      const float* srow = &sm[row][8 + lc];
      float a3 = 0.f, a15 = 0.f;
#pragma unroll
      for (int d = 0; d < 3; ++d)  a3  += srow[d - 1] * w3r[d];
#pragma unroll
      for (int d = 0; d < 15; ++d) a15 += srow[d - 7] * w15r[d];
      float val = (ga * a3 + gb * a15) * osc;
      int n = b * H_ + h;
      int c = cbase + row;
      int l = l0 + lc;
      if (p == 2)
        vbuf[((size_t)n * C_ + c) * L_ + l] = (h16)val;       // (n,c,l)
      else if (p == 1)
        kbuf[((size_t)n * L_ + l) * C_ + c] = (h16)val;       // (n,l,c)
      else
        qbuf[((size_t)n * L_ + l) * C_ + c] = (h16)val;       // (n,l,c)
    }
  }
}

// ---------------- K2: flash attention with async double-buffered LDS ---------
// K tile rows padded to 72 halves, V rows to 40 halves -> conflict-free frags.
__global__ __launch_bounds__(128) void k2_attn(
    const h16* __restrict__ qbuf, const h16* __restrict__ kbuf,
    const h16* __restrict__ vbuf, h16* __restrict__ attT) {
  __shared__ __align__(16) h16 smK[2][32 * 72];
  __shared__ __align__(16) h16 smV[2][64 * 40];
  int tid = threadIdx.x;
  int t = tid & 31, wv = tid >> 5;
  int n = blockIdx.y;
  int q0 = (blockIdx.x * 4 + wv) * 16;
  int b = n / H_, h = n % H_;
  int m = t & 15, ks = t >> 4;

  const h16* kbase = kbuf + (size_t)n * L_ * C_;
  const h16* vbase = vbuf + (size_t)n * C_ * L_;

  // 256 16B chunks of K + 256 of V per stage; 4 copies per thread per stage.
  auto stage_load = [&](int kb, int st) {
    const h16* kg = kbase + (size_t)kb * 32 * C_;
    const h16* vg = vbase + kb * 32;
#pragma unroll
    for (int rep = 0; rep < 2; ++rep) {
      int id = tid + rep * 128;
      cp16(kg + (id >> 3) * C_ + (id & 7) * 8,
           &smK[st][(id >> 3) * 72 + (id & 7) * 8]);
      cp16(vg + (size_t)(id >> 2) * L_ + (id & 3) * 8,
           &smV[st][(id >> 2) * 40 + (id & 3) * 8]);
    }
  };

  const h16* qp = qbuf + ((size_t)n * L_ + q0 + m) * C_ + ks * 16;
  v16h Bq0 = *(const v16h*)qp;
  v16h Bq1 = *(const v16h*)(qp + 32);

  float mr = -3.0e38f, lr = 0.f;
  v8f acc[4] = {{}, {}, {}, {}};

  stage_load(0, 0);
  for (int kb = 0; kb < L_ / 32; ++kb) {
    int cur = kb & 1;
    if (kb + 1 < L_ / 32) {
      stage_load(kb + 1, cur ^ 1);   // overlap next copy with this block's math
      wait_async_le<4>();            // own current-stage asyncs completed
    } else {
      wait_async_le<0>();
    }
    __syncthreads();

    v8f S[2];
#pragma unroll
    for (int s = 0; s < 2; ++s) {
      const h16* kp = &smK[cur][(s * 16 + m) * 72];
      v16h aK0 = a_frag_f16(kp + ks * 16,      kp + 8 + ks * 16);
      v16h aK1 = a_frag_f16(kp + 32 + ks * 16, kp + 40 + ks * 16);
      v8f sacc = {};
      sacc = wmma16(aK0, Bq0, sacc);
      sacc = wmma16(aK1, Bq1, sacc);
      S[s] = sacc;
    }
    // online softmax over k (a column lives in lane pair t / t^16)
    float pm = -3.0e38f;
#pragma unroll
    for (int r = 0; r < 8; ++r) { pm = fmaxf(pm, S[0][r]); pm = fmaxf(pm, S[1][r]); }
    float bm = fmaxf(pm, __shfl_xor(pm, 16));
    float mn = fmaxf(mr, bm);
    float rs = __expf(mr - mn);
    float ex0[8], ex1[8], ps = 0.f;
#pragma unroll
    for (int r = 0; r < 8; ++r) {
      ex0[r] = __expf(S[0][r] - mn);
      ex1[r] = __expf(S[1][r] - mn);
      ps += ex0[r] + ex1[r];
    }
    float bs = ps + __shfl_xor(ps, 16);
    lr = lr * rs + bs;
    mr = mn;
#pragma unroll
    for (int ct = 0; ct < 4; ++ct)
#pragma unroll
      for (int r = 0; r < 8; ++r) acc[ct][r] *= rs;

    // repack exp(S) (C-layout) into B-fragment layout via lane-pair exchange
    unsigned own0[4], own1[4];
#pragma unroll
    for (int j = 0; j < 4; ++j) {
      own0[j] = pack2(ex0[2 * j], ex0[2 * j + 1]);
      own1[j] = pack2(ex1[2 * j], ex1[2 * j + 1]);
    }
    union { v16h v; unsigned u[8]; } P;
#pragma unroll
    for (int j = 0; j < 4; ++j) {
      unsigned s0 = __shfl_xor(own0[j], 16);
      unsigned s1 = __shfl_xor(own1[j], 16);
      P.u[j]     = (ks == 0) ? own0[j] : s1;
      P.u[4 + j] = (ks == 0) ? s0      : own1[j];
    }
    // acc += V * P  (4 c-tiles of 16)
#pragma unroll
    for (int ct = 0; ct < 4; ++ct) {
      const h16* vp = &smV[cur][(ct * 16 + m) * 40];
      v16h aV = a_frag_f16(vp + ks * 16, vp + 8 + ks * 16);
      acc[ct] = wmma16(aV, P.v, acc[ct]);
    }
    __syncthreads();   // all waves done reading before buffer is overwritten
  }
  float inv = 1.f / lr;
  size_t obase = ((size_t)b * L_ + q0 + m) * HC_ + h * C_;
#pragma unroll
  for (int ct = 0; ct < 4; ++ct) {
    v8h o;
#pragma unroll
    for (int r = 0; r < 8; ++r) o[r] = (h16)(acc[ct][r] * inv);
    *(v8h*)(attT + obase + ct * 16 + 8 * ks) = o;   // (b,l,hc), 16B store
  }
}

// ---------------- K3: unify GEMM (64x512) + bias + residual ------------------
__global__ __launch_bounds__(32) void k3_unify(
    const h16* __restrict__ attT, const h16* __restrict__ wu,
    const float* __restrict__ bu, const h16* __restrict__ xqT,
    float* __restrict__ u) {
  int t = threadIdx.x, m = t & 15, ks = t >> 4;
  int lt0 = blockIdx.x * 32;
  int o0  = blockIdx.y * 16;
  int b   = blockIdx.z;
  v8f acc[2] = {{}, {}};
  for (int cc = 0; cc < 16; ++cc) {
    const h16* wr = wu + (size_t)(o0 + m) * HC_ + cc * 32;
    v16h A = a_frag_f16(wr + ks * 16, wr + 8 + ks * 16);
#pragma unroll
    for (int lt = 0; lt < 2; ++lt) {
      const h16* bp = attT + ((size_t)b * L_ + lt0 + lt * 16 + m) * HC_ +
                      cc * 32 + ks * 16;
      acc[lt] = wmma16(A, *(const v16h*)bp, acc[lt]);
    }
  }
#pragma unroll
  for (int lt = 0; lt < 2; ++lt) {
    int l = lt0 + lt * 16 + m;
#pragma unroll
    for (int r = 0; r < 8; ++r) {
      int o = o0 + 8 * ks + r;
      u[((size_t)b * L_ + l) * C_ + o] =
          acc[lt][r] + bu[o] + (float)xqT[((size_t)b * L_ + l) * C_ + o];
    }
  }
}

// ---------------- K4: LayerNorm over channels -> y1T f16, y1f f32 ------------
__global__ void k4_ln(const float* __restrict__ u, const float* __restrict__ g,
                      const float* __restrict__ bt, h16* __restrict__ yT,
                      float* __restrict__ yf) {
  int p = blockIdx.x * blockDim.x + threadIdx.x;   // B*L positions
  const float* x = u + (size_t)p * C_;
  float s = 0.f, s2 = 0.f;
#pragma unroll
  for (int c = 0; c < C_; ++c) { float v = x[c]; s += v; s2 += v * v; }
  float mean = s * (1.f / C_);
  float inv  = rsqrtf(s2 * (1.f / C_) - mean * mean + LN_EPS);
#pragma unroll
  for (int c = 0; c < C_; ++c) {
    float v = (x[c] - mean) * inv * g[c] + bt[c];
    yT[(size_t)p * C_ + c] = (h16)v;
    yf[(size_t)p * C_ + c] = v;
  }
}

// ---------------- K5: FFN1 (256x64) + relu -> hidT f16 -----------------------
__global__ __launch_bounds__(32) void k5_ffn1(
    const h16* __restrict__ yT, const h16* __restrict__ w1,
    const float* __restrict__ b1, h16* __restrict__ hidT) {
  int t = threadIdx.x, m = t & 15, ks = t >> 4;
  int l0 = blockIdx.x * 64;
  int o0 = blockIdx.y * 16;
  int b  = blockIdx.z;
  const h16* wr = w1 + (size_t)(o0 + m) * C_;
  v16h A0 = a_frag_f16(wr + ks * 16,      wr + 8 + ks * 16);
  v16h A1 = a_frag_f16(wr + 32 + ks * 16, wr + 40 + ks * 16);
  v8f acc[4] = {{}, {}, {}, {}};
#pragma unroll
  for (int lt = 0; lt < 4; ++lt) {
    const h16* bp = yT + ((size_t)b * L_ + l0 + lt * 16 + m) * C_ + ks * 16;
    acc[lt] = wmma16(A0, *(const v16h*)bp, acc[lt]);
    acc[lt] = wmma16(A1, *(const v16h*)(bp + 32), acc[lt]);
  }
#pragma unroll
  for (int lt = 0; lt < 4; ++lt) {
    int l = l0 + lt * 16 + m;
    v8h o;
#pragma unroll
    for (int r = 0; r < 8; ++r) {
      float v = acc[lt][r] + b1[o0 + 8 * ks + r];
      o[r] = (h16)fmaxf(v, 0.f);
    }
    *(v8h*)(hidT + ((size_t)b * L_ + l) * 256 + o0 + 8 * ks) = o;
  }
}

// ---------------- K6: FFN2 (64x256) + bias + residual -> u2 ------------------
__global__ __launch_bounds__(32) void k6_ffn2(
    const h16* __restrict__ hidT, const h16* __restrict__ w2,
    const float* __restrict__ b2, const float* __restrict__ yf,
    float* __restrict__ u2) {
  int t = threadIdx.x, m = t & 15, ks = t >> 4;
  int lt0 = blockIdx.x * 32;
  int o0  = blockIdx.y * 16;
  int b   = blockIdx.z;
  v8f acc[2] = {{}, {}};
  for (int cc = 0; cc < 8; ++cc) {
    const h16* wr = w2 + (size_t)(o0 + m) * 256 + cc * 32;
    v16h A = a_frag_f16(wr + ks * 16, wr + 8 + ks * 16);
#pragma unroll
    for (int lt = 0; lt < 2; ++lt) {
      const h16* bp = hidT + ((size_t)b * L_ + lt0 + lt * 16 + m) * 256 +
                      cc * 32 + ks * 16;
      acc[lt] = wmma16(A, *(const v16h*)bp, acc[lt]);
    }
  }
#pragma unroll
  for (int lt = 0; lt < 2; ++lt) {
    int l = lt0 + lt * 16 + m;
#pragma unroll
    for (int r = 0; r < 8; ++r) {
      int o = o0 + 8 * ks + r;
      u2[((size_t)b * L_ + l) * C_ + o] =
          acc[lt][r] + b2[o] + yf[((size_t)b * L_ + l) * C_ + o];
    }
  }
}

// ---------------- K7: LayerNorm + transpose to output (b,c,l) ----------------
__global__ void k7_ln_out(const float* __restrict__ u2, const float* __restrict__ g,
                          const float* __restrict__ bt, float* __restrict__ out) {
  int p = blockIdx.x * blockDim.x + threadIdx.x;   // B*L positions
  int b = p >> 10, l = p & 1023;
  const float* x = u2 + (size_t)p * C_;
  float s = 0.f, s2 = 0.f;
#pragma unroll
  for (int c = 0; c < C_; ++c) { float v = x[c]; s += v; s2 += v * v; }
  float mean = s * (1.f / C_);
  float inv  = rsqrtf(s2 * (1.f / C_) - mean * mean + LN_EPS);
#pragma unroll
  for (int c = 0; c < C_; ++c)
    out[((size_t)b * C_ + c) * L_ + l] = (x[c] - mean) * inv * g[c] + bt[c];
}

extern "C" void kernel_launch(void* const* d_in, const int* in_sizes, int n_in,
                              void* d_out, int out_size, void* d_ws, size_t ws_size,
                              hipStream_t stream) {
  (void)in_sizes; (void)n_in; (void)out_size; (void)ws_size;
  const float* q      = (const float*)d_in[0];
  const float* wq_pw  = (const float*)d_in[1];
  const float* wq_g   = (const float*)d_in[2];
  const float* wq_d3  = (const float*)d_in[3];
  const float* wq_d15 = (const float*)d_in[4];
  const float* wk_pw  = (const float*)d_in[5];
  const float* wk_g   = (const float*)d_in[6];
  const float* wk_d3  = (const float*)d_in[7];
  const float* wk_d15 = (const float*)d_in[8];
  const float* wv_pw  = (const float*)d_in[9];
  const float* wv_g   = (const float*)d_in[10];
  const float* wv_d3  = (const float*)d_in[11];
  const float* wv_d15 = (const float*)d_in[12];
  const float* w_uni  = (const float*)d_in[13];
  const float* b_uni  = (const float*)d_in[14];
  const float* ln1_g  = (const float*)d_in[15];
  const float* ln1_b  = (const float*)d_in[16];
  const float* ln2_g  = (const float*)d_in[17];
  const float* ln2_b  = (const float*)d_in[18];
  const float* ffn_w1 = (const float*)d_in[19];
  const float* ffn_b1 = (const float*)d_in[20];
  const float* ffn_w2 = (const float*)d_in[21];
  const float* ffn_b2 = (const float*)d_in[22];
  float* out = (float*)d_out;

  char* w = (char*)d_ws;
  auto take = [&](size_t bytes) {
    char* p = w;
    w += (bytes + 255) & ~(size_t)255;
    return p;
  };
  const size_t NL = (size_t)B_ * L_;
  h16*   xqT  = (h16*)take(NL * C_ * 2);                    // (b,l,c) f16
  h16*   qbuf = (h16*)take((size_t)B_ * H_ * L_ * C_ * 2);  // (n,l,c)
  h16*   kbuf = (h16*)take((size_t)B_ * H_ * L_ * C_ * 2);  // (n,l,c)
  h16*   vbuf = (h16*)take((size_t)B_ * H_ * C_ * L_ * 2);  // (n,c,l)
  h16*   attT = (h16*)take(NL * HC_ * 2);                   // (b,l,hc)
  float* ubuf = (float*)take(NL * C_ * 4);                  // (b,l,c)
  h16*   y1T  = (h16*)take(NL * C_ * 2);                    // (b,l,c)
  float* y1f  = (float*)take(NL * C_ * 4);                  // (b,l,c)
  h16*   hidT = (h16*)take(NL * 256 * 2);                   // (b,l,4c)
  float* u2   = (float*)take(NL * C_ * 4);                  // (b,l,c)
  h16*   pwq16 = (h16*)take((size_t)HC_ * C_ * 2);
  h16*   pwk16 = (h16*)take((size_t)HC_ * C_ * 2);
  h16*   pwv16 = (h16*)take((size_t)HC_ * C_ * 2);
  h16*   wu16  = (h16*)take((size_t)C_ * HC_ * 2);
  h16*   w116  = (h16*)take((size_t)256 * C_ * 2);
  h16*   w216  = (h16*)take((size_t)C_ * 256 * 2);

  // weight pre-conversion (tiny, once)
  kcvt<<<dim3(128), dim3(256), 0, stream>>>(wq_pw, pwq16, HC_ * C_);
  kcvt<<<dim3(128), dim3(256), 0, stream>>>(wk_pw, pwk16, HC_ * C_);
  kcvt<<<dim3(128), dim3(256), 0, stream>>>(wv_pw, pwv16, HC_ * C_);
  kcvt<<<dim3(128), dim3(256), 0, stream>>>(w_uni, wu16, C_ * HC_);
  kcvt<<<dim3(64),  dim3(256), 0, stream>>>(ffn_w1, w116, 256 * C_);
  kcvt<<<dim3(64),  dim3(256), 0, stream>>>(ffn_w2, w216, C_ * 256);

  k0_transpose<<<dim3((B_ * C_ * L_) / 256), dim3(256), 0, stream>>>(q, xqT);

  k1_qkv<<<dim3(L_ / TL_, HC_ / 16, B_ * 3), dim3(32), 0, stream>>>(
      xqT, pwq16, wq_g, wq_d3, wq_d15, pwk16, wk_g, wk_d3, wk_d15,
      pwv16, wv_g, wv_d3, wv_d15, qbuf, kbuf, vbuf);

  k2_attn<<<dim3(L_ / 64, B_ * H_), dim3(128), 0, stream>>>(qbuf, kbuf, vbuf, attT);

  k3_unify<<<dim3(L_ / 32, C_ / 16, B_), dim3(32), 0, stream>>>(
      attT, wu16, b_uni, xqT, ubuf);

  k4_ln<<<dim3((B_ * L_) / 256), dim3(256), 0, stream>>>(ubuf, ln1_g, ln1_b, y1T, y1f);

  k5_ffn1<<<dim3(L_ / 64, 256 / 16, B_), dim3(32), 0, stream>>>(y1T, w116, ffn_b1, hidT);

  k6_ffn2<<<dim3(L_ / 32, C_ / 16, B_), dim3(32), 0, stream>>>(hidT, w216, ffn_b2, y1f, u2);

  k7_ln_out<<<dim3((B_ * L_) / 256), dim3(256), 0, stream>>>(u2, ln2_g, ln2_b, out);
}
